// YOLOLoss_39041252721250
// MI455X (gfx1250) — compile-verified, matrix operand order checked by reading, not run on hardware
//
#include <hip/hip_runtime.h>

namespace {
constexpr int Bn  = 8;
constexpr int An  = 21504;            // 128^2 + 64^2 + 32^2
constexpr int Gn  = 20;
constexpr int Cn  = 3;
constexpr int IMG = 1024;
constexpr long long IMG2 = (long long)IMG * IMG;

// workspace layout (units: float)
constexpr size_t OFF_BOX  = 0;                                  // [B*A*4]
constexpr size_t OFF_OBJ  = OFF_BOX  + (size_t)4 * Bn * An;     // [B*A]
constexpr size_t OFF_CLS  = OFF_OBJ  + (size_t)Bn * An;         // [B*A*3]
constexpr size_t OFF_BASE = OFF_CLS  + (size_t)3 * Bn * An;     // [B*A]
constexpr size_t OFF_CTR  = OFF_BASE + (size_t)Bn * An;         // [B*A*2]
constexpr size_t OFF_DH   = OFF_CTR  + (size_t)2 * Bn * An;     // _Float16[4] per anchor
constexpr size_t OFF_COST = OFF_DH   + (size_t)2 * Bn * An;     // [B*G*A]
constexpr size_t OFF_CAND = OFF_COST + (size_t)Bn * Gn * An;    // [B*A]
constexpr size_t OFF_THR  = OFF_CAND + (size_t)Bn * An;         // [B*G]
constexpr size_t OFF_ACC  = OFF_THR  + (size_t)Bn * Gn;         // [16] accumulators
} // namespace

typedef _Float16 v16h __attribute__((ext_vector_type(16)));
typedef float    v8f  __attribute__((ext_vector_type(8)));

__device__ __forceinline__ float sigf(float x) { return 1.f / (1.f + __expf(-x)); }

__device__ __forceinline__ float bcef(float x, float y) {
  return fmaxf(x, 0.f) - x * y + log1pf(__expf(-fabsf(x)));
}

__device__ __forceinline__ float iou_xywh(float gx, float gy, float gw, float gh,
                                          float px, float py, float pw, float ph) {
  float tlx = fmaxf(gx - gw * 0.5f, px - pw * 0.5f);
  float tly = fmaxf(gy - gh * 0.5f, py - ph * 0.5f);
  float brx = fminf(gx + gw * 0.5f, px + pw * 0.5f);
  float bry = fminf(gy + gh * 0.5f, py + ph * 0.5f);
  float en = (tlx < brx && tly < bry) ? 1.f : 0.f;
  float ai = (brx - tlx) * (bry - tly) * en;
  float au = gw * gh + pw * ph - ai;
  return ai / (au + 1e-16f);
}

__device__ __forceinline__ void block_sum_atomic(float v, float* dst, float* sbuf) {
  int tid = threadIdx.x;
  sbuf[tid] = v;
  __syncthreads();
  for (int s = blockDim.x >> 1; s > 0; s >>= 1) {
    if (tid < s) sbuf[tid] += sbuf[tid + s];
    __syncthreads();
  }
  if (tid == 0) atomicAdd(dst, sbuf[0]);
  __syncthreads();
}

// ---------------- kernel 0: zero accumulators ----------------
__global__ void k_init(float* ws) {
  if (threadIdx.x < 16) (ws + OFF_ACC)[threadIdx.x] = 0.f;
}

// ---------------- kernel 1: decode anchors ----------------
__global__ void k_decode(const float* __restrict__ p3, const float* __restrict__ p4,
                         const float* __restrict__ p5, float* __restrict__ ws) {
  int idx = blockIdx.x * blockDim.x + threadIdx.x;
  if (idx >= Bn * An) return;
  int b = idx / An, a = idx % An;
  const float* src; int off, H, s;
  if (a < 16384)      { src = p3; off = 0;     H = 128; s = 8;  }
  else if (a < 20480) { src = p4; off = 16384; H = 64;  s = 16; }
  else                { src = p5; off = 20480; H = 32;  s = 32; }
  int local = a - off, HW = H * H;
  int x = local % H, y = local / H;
  const float* q = src + (size_t)b * 8 * HW + local;
  float o[8];
#pragma unroll
  for (int c = 0; c < 8; ++c) o[c] = q[(size_t)c * HW];

  size_t ai = (size_t)idx;
  float* box  = ws + OFF_BOX;
  float* obj  = ws + OFF_OBJ;
  float* cls  = ws + OFF_CLS;
  float* basep = ws + OFF_BASE;
  float* ctr  = ws + OFF_CTR;
  _Float16* dh = ((_Float16*)(ws + OFF_DH)) + ai * 4;

  box[ai * 4 + 0] = (o[0] + x) * (float)s;
  box[ai * 4 + 1] = (o[1] + y) * (float)s;
  box[ai * 4 + 2] = __expf(o[2]) * (float)s;
  box[ai * 4 + 3] = __expf(o[3]) * (float)s;
  obj[ai] = o[4];
  float so = sigf(o[4]);
  float basev = 0.f;
#pragma unroll
  for (int c = 0; c < Cn; ++c) {
    float xv = o[5 + c];
    cls[ai * 3 + c] = xv;
    float p = sqrtf(sigf(xv) * so);
    basev += __logf(1.f - p + 1e-8f);
    dh[c] = (_Float16)(__logf(p + 1e-8f) - __logf(1.f - p + 1e-8f));
  }
  dh[3] = (_Float16)0.f;
  basep[ai] = basev;
  ctr[ai * 2 + 0] = (x + 0.5f) * (float)s;
  ctr[ai * 2 + 1] = (y + 0.5f) * (float)s;
}

// ---------------- kernel 2: cost matrix with WMMA class-cost einsum ----------------
// one wave (32 lanes) per (16-anchor tile, batch). D = onehot(G x 32pad) x d^T(32 x 16 anchors)
__global__ void __launch_bounds__(32) k_cost(const float* __restrict__ labels,
                                             float* __restrict__ ws) {
  const int lane = threadIdx.x;
  const int b = blockIdx.y;
  const int a0 = blockIdx.x * 16;
  const int a = a0 + (lane & 15);
  const size_t ai = (size_t)b * An + a;

  const float* box   = ws + OFF_BOX;
  const float* ctr   = ws + OFF_CTR;
  const float* basep = ws + OFF_BASE;
  const _Float16* dhp = (const _Float16*)(ws + OFF_DH);
  float* cost  = ws + OFF_COST;
  float* candm = ws + OFF_CAND;

  // A-matrix (16-bit, 16x32): lanes 0-15 hold M=lane, K=0..7 in halves 0..7 (only K<3 nonzero)
  // B-matrix (16-bit, 32x16): lanes 0-15 hold N=lane, K=0..7 in halves 0..7
  v16h Am0 = {}; v16h Am1 = {}; v16h Bm = {};
  if (lane < 16) {
    int c0 = (int)labels[((size_t)b * Gn + lane) * 5 + 4];
    if (c0 >= 0 && c0 < Cn) Am0[c0] = (_Float16)1.0f;
    int g1 = 16 + lane;
    if (g1 < Gn) {
      int c1 = (int)labels[((size_t)b * Gn + g1) * 5 + 4];
      if (c1 >= 0 && c1 < Cn) Am1[c1] = (_Float16)1.0f;
    }
    const _Float16* dh = dhp + ai * 4;
    Bm[0] = dh[0]; Bm[1] = dh[1]; Bm[2] = dh[2];
  }
  v8f C0 = {}; v8f C1 = {};
  C0 = __builtin_amdgcn_wmma_f32_16x16x32_f16(false, Am0, false, Bm, (short)0, C0, false, false);
  C1 = __builtin_amdgcn_wmma_f32_16x16x32_f16(false, Am1, false, Bm, (short)0, C1, false, false);

  float px = box[ai * 4 + 0], py = box[ai * 4 + 1];
  float pw = box[ai * 4 + 2], ph = box[ai * 4 + 3];
  float xc = ctr[ai * 2 + 0], yc = ctr[ai * 2 + 1];
  float sfl = (a < 16384) ? 8.f : (a < 20480) ? 16.f : 32.f;
  float basev = basep[ai];

  // pass 1: geometric candidate flags for the (g,a) pairs this lane owns
  unsigned candAny = 0, ibMask = 0, icMask = 0;
  for (int t = 0; t < 2; ++t) {
#pragma unroll
    for (int r = 0; r < 8; ++r) {
      int g = r + ((lane < 16) ? 0 : 8) + t * 16;
      if (g >= Gn) continue;
      const float* gt = labels + ((size_t)b * Gn + g) * 5;
      float gx = gt[0], gy = gt[1], gw = gt[2], gh = gt[3];
      bool ib = (xc > gx - gw * 0.5f) && (xc < gx + gw * 0.5f) &&
                (yc > gy - gh * 0.5f) && (yc < gy + gh * 0.5f);
      float rr = 2.5f * sfl;
      bool ic = (xc > gx - rr) && (xc < gx + rr) && (yc > gy - rr) && (yc < gy + rr);
      int bit = t * 8 + r;
      if (ib) ibMask |= 1u << bit;
      if (ic) icMask |= 1u << bit;
      if (ib || ic) candAny = 1u;
    }
  }
  // lanes l and l^16 cover the complementary GT rows of the same anchor column
  candAny |= (unsigned)__shfl_xor((int)candAny, 16, 32);
  if (lane < 16) candm[ai] = candAny ? 1.f : 0.f;

  // pass 2: finalize cost entries (einsum value from WMMA accumulators)
  for (int t = 0; t < 2; ++t) {
#pragma unroll
    for (int r = 0; r < 8; ++r) {
      int g = r + ((lane < 16) ? 0 : 8) + t * 16;
      if (g >= Gn) continue;
      const float* gt = labels + ((size_t)b * Gn + g) * 5;
      float iou = iou_xywh(gt[0], gt[1], gt[2], gt[3], px, py, pw, ph);
      float iou_m = candAny ? iou : 0.f;
      float iou_cost = -__logf(iou_m + 1e-8f);
      float eins = (t == 0) ? C0[r] : C1[r];
      float cls_cost = -basev - eins;
      int bit = t * 8 + r;
      bool inboth = ((ibMask >> bit) & 1u) && ((icMask >> bit) & 1u);
      float c = cls_cost + 3.f * iou_cost + (inboth ? 0.f : 100000.f);
      cost[((size_t)b * Gn + g) * An + a] = candAny ? c : 1e9f;
    }
  }
}

// ---------------- kernel 3: per-(b,g) dyn_k + k-th smallest cost ----------------
__global__ void __launch_bounds__(256) k_topk(const float* __restrict__ labels,
                                              float* __restrict__ ws) {
  const int g = blockIdx.x, b = blockIdx.y, tid = threadIdx.x;
  const float* box   = ws + OFF_BOX;
  const float* candm = ws + OFF_CAND;
  const float* cost  = ws + OFF_COST + ((size_t)b * Gn + g) * An;
  float* thr = ws + OFF_THR;

  const float* gt = labels + ((size_t)b * Gn + g) * 5;
  float gx = gt[0], gy = gt[1], gw = gt[2], gh = gt[3];

  float tmin[10], tmax[10];
#pragma unroll
  for (int k = 0; k < 10; ++k) { tmin[k] = 1e30f; tmax[k] = -1e30f; }

  for (int a = tid; a < An; a += 256) {
    size_t ai = (size_t)b * An + a;
    float cnd = candm[ai];
    float iou = 0.f;
    if (cnd > 0.5f)
      iou = iou_xywh(gx, gy, gw, gh, box[ai * 4 + 0], box[ai * 4 + 1],
                     box[ai * 4 + 2], box[ai * 4 + 3]);
    float c = cost[a];
    if (c < tmin[9]) {            // insert ascending
      int k = 9;
      while (k > 0 && tmin[k - 1] > c) { tmin[k] = tmin[k - 1]; --k; }
      tmin[k] = c;
    }
    if (iou > tmax[9]) {          // insert descending
      int k = 9;
      while (k > 0 && tmax[k - 1] < iou) { tmax[k] = tmax[k - 1]; --k; }
      tmax[k] = iou;
    }
  }

  __shared__ float smin[256][10];
  __shared__ float smax[256][10];
#pragma unroll
  for (int k = 0; k < 10; ++k) { smin[tid][k] = tmin[k]; smax[tid][k] = tmax[k]; }
  __syncthreads();

  for (int s = 128; s > 0; s >>= 1) {
    if (tid < s) {
      float oa[10], ob[10];
      int i = 0, j = 0;
#pragma unroll
      for (int k = 0; k < 10; ++k) {
        float av = smin[tid][i], bv = smin[tid + s][j];
        if (av <= bv) { oa[k] = av; ++i; } else { oa[k] = bv; ++j; }
      }
      i = 0; j = 0;
#pragma unroll
      for (int k = 0; k < 10; ++k) {
        float av = smax[tid][i], bv = smax[tid + s][j];
        if (av >= bv) { ob[k] = av; ++i; } else { ob[k] = bv; ++j; }
      }
#pragma unroll
      for (int k = 0; k < 10; ++k) { smin[tid][k] = oa[k]; smax[tid][k] = ob[k]; }
    }
    __syncthreads();
  }
  if (tid == 0) {
    float su = 0.f;
#pragma unroll
    for (int k = 0; k < 10; ++k) su += fmaxf(smax[0][k], 0.f);
    int dynk = (int)su; dynk = (dynk < 1) ? 1 : (dynk > 10 ? 10 : dynk);
    thr[b * Gn + g] = smin[0][dynk - 1];
  }
}

// ---------------- kernel 4: resolve matches, detection losses ----------------
__global__ void __launch_bounds__(256) k_anchor(const float* __restrict__ labels,
                                                float* __restrict__ ws) {
  int idx = blockIdx.x * blockDim.x + threadIdx.x;   // grid exactly covers Bn*An
  int b = idx / An, a = idx % An;
  size_t ai = (size_t)idx;
  const float* box   = ws + OFF_BOX;
  const float* objp  = ws + OFF_OBJ;
  const float* clsp  = ws + OFF_CLS;
  const float* candm = ws + OFF_CAND;
  const float* cost  = ws + OFF_COST;
  const float* thr   = ws + OFF_THR;
  float* acc = ws + OFF_ACC;

  float px = box[ai * 4 + 0], py = box[ai * 4 + 1];
  float pw = box[ai * 4 + 2], ph = box[ai * 4 + 3];

  int nm = 0, gfirst = 0, gmin = 0;
  float cmin = 3e38f;
  if (candm[ai] > 0.5f) {
    for (int g = 0; g < Gn; ++g) {
      float c = cost[((size_t)b * Gn + g) * An + a];
      if (c < cmin) { cmin = c; gmin = g; }
      if (c <= thr[b * Gn + g]) { if (nm == 0) gfirst = g; ++nm; }
    }
  }
  int matched = (nm > 1) ? gmin : gfirst;
  float fg = (nm > 0) ? 1.f : 0.f;

  float liou = 0.f, lcls = 0.f;
  if (nm > 0) {
    const float* gt = labels + ((size_t)b * Gn + matched) * 5;
    float iou = iou_xywh(gt[0], gt[1], gt[2], gt[3], px, py, pw, ph);
    liou = 1.f - iou * iou;
    int gc = (int)gt[4];
#pragma unroll
    for (int c = 0; c < Cn; ++c) {
      float tgt = (c == gc) ? iou : 0.f;
      lcls += bcef(clsp[ai * 3 + c], tgt);
    }
  }
  float lobj = bcef(objp[ai], fg);

  __shared__ float sbuf[256];
  block_sum_atomic(fg,   acc + 0, sbuf);
  block_sum_atomic(liou, acc + 1, sbuf);
  block_sum_atomic(lobj, acc + 2, sbuf);
  block_sum_atomic(lcls, acc + 3, sbuf);
}

// ---------------- kernel 5: mask CE + dice partials (bandwidth-bound) ----------------
__global__ void __launch_bounds__(256) k_mask(const float* __restrict__ mi,
                                              const int* __restrict__ ml,
                                              float* __restrict__ ws) {
  float* acc = ws + OFF_ACC;
  long long nvec = (long long)Bn * IMG2 / 4;
  long long step = (long long)gridDim.x * blockDim.x;
  float ce = 0.f, tp0 = 0.f, tp1 = 0.f, tp2 = 0.f;
  float sm0 = 0.f, sm1 = 0.f, sm2 = 0.f, oh0 = 0.f, oh1 = 0.f, oh2 = 0.f;

  for (long long v = (long long)blockIdx.x * blockDim.x + threadIdx.x; v < nvec; v += step) {
    long long p = v * 4;
    int b = (int)(p / IMG2);
    long long pix = p % IMG2;
    const float* base = mi + (size_t)b * 3 * IMG2 + pix;
    __builtin_prefetch(base + 4 * step, 0, 1);        // global_prefetch_b8
    float4 x0 = *reinterpret_cast<const float4*>(base);
    float4 x1 = *reinterpret_cast<const float4*>(base + IMG2);
    float4 x2 = *reinterpret_cast<const float4*>(base + 2 * IMG2);
    int4 lb = *reinterpret_cast<const int4*>(ml + (size_t)b * IMG2 + pix);
    float a0v[4] = {x0.x, x0.y, x0.z, x0.w};
    float a1v[4] = {x1.x, x1.y, x1.z, x1.w};
    float a2v[4] = {x2.x, x2.y, x2.z, x2.w};
    int lbv[4] = {lb.x, lb.y, lb.z, lb.w};
#pragma unroll
    for (int j = 0; j < 4; ++j) {
      float v0 = a0v[j], v1 = a1v[j], v2 = a2v[j];
      float m = fmaxf(v0, fmaxf(v1, v2));
      float e0 = __expf(v0 - m), e1 = __expf(v1 - m), e2 = __expf(v2 - m);
      float sum = e0 + e1 + e2;
      float inv = 1.f / sum;
      float lse = m + __logf(sum);
      float s0 = e0 * inv, s1 = e1 * inv, s2 = e2 * inv;
      sm0 += s0; sm1 += s1; sm2 += s2;
      int lab = lbv[j];
      float xl = (lab == 0) ? v0 : (lab == 1) ? v1 : v2;
      ce += (lse - xl);
      if (lab == 0)      { tp0 += s0; oh0 += 1.f; }
      else if (lab == 1) { tp1 += s1; oh1 += 1.f; }
      else if (lab == 2) { tp2 += s2; oh2 += 1.f; }
    }
  }

  __shared__ float sbuf[256];
  block_sum_atomic(ce,  acc + 4,  sbuf);
  block_sum_atomic(tp0, acc + 5,  sbuf);
  block_sum_atomic(tp1, acc + 6,  sbuf);
  block_sum_atomic(tp2, acc + 7,  sbuf);
  block_sum_atomic(sm0, acc + 8,  sbuf);
  block_sum_atomic(sm1, acc + 9,  sbuf);
  block_sum_atomic(sm2, acc + 10, sbuf);
  block_sum_atomic(oh0, acc + 11, sbuf);
  block_sum_atomic(oh1, acc + 12, sbuf);
  block_sum_atomic(oh2, acc + 13, sbuf);
}

// ---------------- kernel 6: finalize ----------------
__global__ void k_final(const float* __restrict__ ws, float* __restrict__ out) {
  const float* acc = ws + OFF_ACC;
  float nfg = fmaxf(acc[0], 1.f);
  float ce = acc[4] / (float)((long long)Bn * IMG2);
  float d = 0.f;
#pragma unroll
  for (int c = 0; c < Cn; ++c) {
    float tp = acc[5 + c];
    float fp = acc[8 + c] - tp;
    float fn = acc[11 + c] - tp;
    d += (2.f * tp + 1e-5f) / (2.f * tp + fn + fp + 1e-5f);
  }
  float dice = 1.f - d / (float)Cn;
  float lmask = ce + dice;
  float liou = 5.f * acc[1] / nfg;
  float lobj = acc[2] / nfg;
  float lcls = acc[3] / nfg;
  out[0] = liou + lobj + lcls + lmask;
  out[1] = liou;
  out[2] = lobj;
  out[3] = lcls;
  out[4] = lmask;
  out[5] = 0.f;
}

extern "C" void kernel_launch(void* const* d_in, const int* in_sizes, int n_in,
                              void* d_out, int out_size, void* d_ws, size_t ws_size,
                              hipStream_t stream) {
  const float* p3     = (const float*)d_in[0];
  const float* p4     = (const float*)d_in[1];
  const float* p5     = (const float*)d_in[2];
  const float* mi     = (const float*)d_in[3];
  const float* labels = (const float*)d_in[4];
  const int*   ml     = (const int*)d_in[6];
  float* out = (float*)d_out;
  float* ws  = (float*)d_ws;

  k_init  <<<1, 64, 0, stream>>>(ws);
  k_decode<<<(Bn * An + 255) / 256, 256, 0, stream>>>(p3, p4, p5, ws);
  k_cost  <<<dim3(An / 16, Bn), 32, 0, stream>>>(labels, ws);
  k_topk  <<<dim3(Gn, Bn), 256, 0, stream>>>(labels, ws);
  k_anchor<<<(Bn * An) / 256, 256, 0, stream>>>(labels, ws);
  k_mask  <<<2048, 256, 0, stream>>>(mi, ml, ws);
  k_final <<<1, 1, 0, stream>>>(ws, out);
}
